// ScaledDotProductAttention_86809878987108
// MI455X (gfx1250) — matrix operand order repaired and checked by reading, not compiled, not run on hardware
//
#include <hip/hip_runtime.h>
#include <hip/hip_bf16.h>
#include <math.h>

typedef __bf16 bf16;
typedef __bf16 v16bf __attribute__((ext_vector_type(16)));
typedef float  v8f   __attribute__((ext_vector_type(8)));

#define WMMA_BF16(A, B, C) \
  __builtin_amdgcn_wmma_f32_16x16x32_bf16(false, (A), false, (B), (short)0, (C), false, false)

static __device__ __forceinline__ void split_bf16(float x, bf16& h, bf16& l) {
  h = (bf16)x;
  l = (bf16)(x - (float)h);
}

constexpr int   B_   = 16;
constexpr int   LQ   = 2048;
constexpr int   LK   = 2048;
constexpr int   DH   = 64;
constexpr int   QT   = 16;            // q-rows per workgroup
constexpr int   NW   = 8;             // waves per workgroup
constexpr int   KSL  = LK / NW;       // 256 k-columns per wave
constexpr int   SP   = LK + 4;        // padded LDS row stride (2052, mult of 4 -> b128 ok)
constexpr int   QSP  = 68;            // Qs stride: mult of 4, 68%64==4 -> conflict-free columns
constexpr float NEG_BIG = -65500.0f;
constexpr size_t NKV = (size_t)B_ * LK * DH;

// ---- preprocess: fp32 -> bf16 hi/lo split of K (row-major) and V (transposed) ----
__global__ __launch_bounds__(256)
void preprocess_kernel(const float* __restrict__ Kp, const float* __restrict__ Vp,
                       bf16* __restrict__ Khi, bf16* __restrict__ Klo,
                       bf16* __restrict__ Vthi, bf16* __restrict__ Vtlo) {
  __shared__ float Vs[64 * 65];
  const int tid = threadIdx.x;
  const int b   = blockIdx.x >> 5;
  const int k0  = (blockIdx.x & 31) * 64;
  for (int i = tid; i < 64 * 64; i += 256) {
    int k = i >> 6, d = i & 63;
    size_t idx = ((size_t)b * LK + k0 + k) * DH + d;
    float x = Kp[idx];
    bf16 h, l; split_bf16(x, h, l);
    Khi[idx] = h; Klo[idx] = l;
  }
  for (int i = tid; i < 64 * 64; i += 256) {
    int k = i >> 6, d = i & 63;
    Vs[k * 65 + d] = Vp[((size_t)b * LK + k0 + k) * DH + d];
  }
  __syncthreads();
  for (int i = tid; i < 64 * 64; i += 256) {
    int d = i >> 6, k = i & 63;
    float x = Vs[k * 65 + d];
    bf16 h, l; split_bf16(x, h, l);
    size_t o = ((size_t)b * DH + d) * LK + k0 + k;
    Vthi[o] = h; Vtlo[o] = l;
  }
}

template <bool PRE>
__global__ __launch_bounds__(256)
void attn_fused_kernel(const float* __restrict__ Qp,
                       const float* __restrict__ Kp,
                       const float* __restrict__ Vp,
                       const int*   __restrict__ Mp,
                       const float* __restrict__ Tp,
                       const bf16*  __restrict__ Khi, const bf16* __restrict__ Klo,
                       const bf16*  __restrict__ Vthi, const bf16* __restrict__ Vtlo,
                       float* __restrict__ Out) {
  __shared__ float S[QT * SP];                         // scores -> exp(s - max) (unnormalized)
  __shared__ float Qs[QT * QSP];
  __shared__ float Kst[PRE ? 4 : NW * 16 * 33];        // fallback-only staging
  __shared__ float Ts[QT];
  __shared__ float rowred[QT * NW];
  __shared__ float rowmax[QT];
  __shared__ float rowinv[QT];
  __shared__ float ctx[QT * DH];

  const int tid  = threadIdx.x;
  const int lane = tid & 31;
  const int w    = tid >> 5;
  const int blk  = blockIdx.x;
  const int b    = blk >> 7;
  const int q0   = (blk & 127) * QT;

  const int hl  = (lane < 16) ? 0 : 1;
  const int l15 = lane & 15;

  // ---- phase 0: stage Q tile (float4) + T ----
  {
    const float* Qg = Qp + ((size_t)b * LQ + q0) * DH;
    {
      int r = tid >> 4, d4 = (tid & 15) * 4;
      *(float4*)&Qs[r * QSP + d4] = *(const float4*)&Qg[r * DH + d4];
    }
    if (tid < QT) Ts[tid] = Tp[(size_t)b * LQ + q0 + tid];
  }
  __syncthreads();

  // ---- Q A-tiles bf16 hi/lo ----
  v16bf aqh[2], aql[2];
  {
    const int koff = hl ? 16 : 0;
#pragma unroll
    for (int h = 0; h < 2; ++h) {
#pragma unroll
      for (int e = 0; e < 16; ++e) {
        int d = h * 32 + koff + e;
        float x = Qs[l15 * QSP + d];
        bf16 hi, lo; split_bf16(x, hi, lo);
        aqh[h][e] = hi; aql[h][e] = lo;
      }
    }
  }
  float tinv[8];
#pragma unroll
  for (int v = 0; v < 8; ++v)
    tinv[v] = 0.125f / Ts[v + (hl ? 8 : 0)];

  // ---- phase 1: S = mask((Q K^T) * scale / T); fused row-max tracking ----
  float lmax[8];
#pragma unroll
  for (int v = 0; v < 8; ++v) lmax[v] = -3.0e38f;
  {
    const int* mbase = Mp + ((size_t)b * LQ + q0) * LK + l15;
    const int  eoff  = hl ? 16 : 0;
    for (int kt = 0; kt < KSL / 16; ++kt) {
      const int k0 = w * KSL + kt * 16;
      v8f acc = {};
      if constexpr (PRE) {
        const bf16* kh = Khi + ((size_t)b * LK + k0 + l15) * DH + eoff;
        const bf16* kl = Klo + ((size_t)b * LK + k0 + l15) * DH + eoff;
#pragma unroll
        for (int h = 0; h < 2; ++h) {
          v16bf bh = *(const v16bf*)(kh + h * 32);
          v16bf bl = *(const v16bf*)(kl + h * 32);
          acc = WMMA_BF16(aqh[h], bh, acc);
          acc = WMMA_BF16(aqh[h], bl, acc);
          acc = WMMA_BF16(aql[h], bh, acc);
        }
      } else {
#pragma unroll
        for (int h = 0; h < 2; ++h) {
#pragma unroll
          for (int i = 0; i < 16; ++i)
            Kst[w * (16 * 33) + i * 33 + lane] =
                Kp[((size_t)b * LK + k0 + i) * DH + h * 32 + lane];
          asm volatile("s_wait_dscnt 0" ::: "memory");
          v16bf bh, bl;
#pragma unroll
          for (int e = 0; e < 16; ++e) {
            float x = Kst[w * (16 * 33) + l15 * 33 + eoff + e];
            bf16 hi, lo; split_bf16(x, hi, lo);
            bh[e] = hi; bl[e] = lo;
          }
          acc = WMMA_BF16(aqh[h], bh, acc);
          acc = WMMA_BF16(aqh[h], bl, acc);
          acc = WMMA_BF16(aql[h], bh, acc);
        }
      }
      if (kt + 1 < KSL / 16)
        __builtin_prefetch(mbase + (size_t)(hl ? 8 : 0) * LK + k0 + 16, 0, 1);
#pragma unroll
      for (int v = 0; v < 8; ++v) {
        int M = v + (hl ? 8 : 0);
        int m = mbase[(size_t)M * LK + k0];
        float sv = acc[v] * tinv[v];
        sv = (m == 0) ? NEG_BIG : sv;
        lmax[v] = fmaxf(lmax[v], sv);
        S[M * SP + k0 + l15] = sv;
      }
    }
#pragma unroll
    for (int v = 0; v < 8; ++v) {
      float lm = lmax[v];
#pragma unroll
      for (int off = 1; off < 16; off <<= 1) lm = fmaxf(lm, __shfl_xor(lm, off));
      if (l15 == 0) rowred[(v + (hl ? 8 : 0)) * NW + w] = lm;
    }
  }
  __syncthreads();
  if (tid < QT) {
    float m = rowred[tid * NW];
    for (int i = 1; i < NW; ++i) m = fmaxf(m, rowred[tid * NW + i]);
    rowmax[tid] = m;
  }
  __syncthreads();

  // ---- phase B: exp + row sum (float4 LDS passes) ----
  for (int r = 0; r < QT; ++r) {
    float mx = rowmax[r];
    float ls = 0.f;
    for (int c4 = tid * 4; c4 < LK; c4 += 1024) {
      float4 x = *(float4*)&S[r * SP + c4];
      x.x = __expf(x.x - mx);
      x.y = __expf(x.y - mx);
      x.z = __expf(x.z - mx);
      x.w = __expf(x.w - mx);
      *(float4*)&S[r * SP + c4] = x;
      ls += (x.x + x.y) + (x.z + x.w);
    }
#pragma unroll
    for (int off = 16; off > 0; off >>= 1) ls += __shfl_xor(ls, off);
    if (lane == 0) rowred[r * NW + w] = ls;
  }
  __syncthreads();
  if (tid < QT) {
    float s = 0.f;
    for (int i = 0; i < NW; ++i) s += rowred[tid * NW + i];
    rowinv[tid] = 1.0f / s;
  }
  __syncthreads();

  // ---- phase C: normalize on the fly, stream attention out (b128); zero ctx ----
  {
    float* attg = Out + (size_t)B_ * LQ * DH + ((size_t)b * LQ + q0) * LK;
    for (int r = 0; r < QT; ++r) {
      float ri = rowinv[r];
      for (int c4 = tid * 4; c4 < LK; c4 += 1024) {
        float4 x = *(const float4*)&S[r * SP + c4];
        x.x *= ri; x.y *= ri; x.z *= ri; x.w *= ri;
        *(float4*)&attg[(size_t)r * LK + c4] = x;   // S keeps unnormalized exp; D folds rowinv
      }
    }
    if (tid < QT * DH / 4) *(float4*)&ctx[tid * 4] = float4{0.f, 0.f, 0.f, 0.f};
  }
  __syncthreads();

  // ---- phase D: context = (P~ * rowinv) @ V  (bf16x3 WMMA) ----
  {
    const int koffA = hl ? 16 : 0;
    const int eoffB = hl ? 16 : 0;
    const float ri = rowinv[l15];                 // A-tile row for this lane
    v8f acc[4] = {{}, {}, {}, {}};
    for (int kc = 0; kc < KSL / 32; ++kc) {
      const int k0 = w * KSL + kc * 32;
      const int abase = l15 * SP + k0 + koffA;    // 16B aligned
      float4 g0 = *(const float4*)&S[abase];
      float4 g1 = *(const float4*)&S[abase + 4];
      float4 g2 = *(const float4*)&S[abase + 8];
      float4 g3 = *(const float4*)&S[abase + 12];
      float xs[16] = {g0.x, g0.y, g0.z, g0.w, g1.x, g1.y, g1.z, g1.w,
                      g2.x, g2.y, g2.z, g2.w, g3.x, g3.y, g3.z, g3.w};
      v16bf pah, pal;
#pragma unroll
      for (int e = 0; e < 16; ++e) {
        float x = xs[e] * ri;
        bf16 hi, lo; split_bf16(x, hi, lo);
        pah[e] = hi; pal[e] = lo;
      }
#pragma unroll
      for (int dt = 0; dt < 4; ++dt) {
        v16bf bh, bl;
        if constexpr (PRE) {
          size_t o = ((size_t)b * DH + dt * 16 + l15) * LK + k0 + eoffB;
          bh = *(const v16bf*)(Vthi + o);
          bl = *(const v16bf*)(Vtlo + o);
        } else {
          const float* vb = Vp + ((size_t)b * LK + k0 + eoffB) * DH + dt * 16 + l15;
#pragma unroll
          for (int e = 0; e < 16; ++e) {
            float x = vb[(size_t)e * DH];
            bf16 hi, lo; split_bf16(x, hi, lo);
            bh[e] = hi; bl[e] = lo;
          }
        }
        acc[dt] = WMMA_BF16(pah, bh, acc[dt]);
        acc[dt] = WMMA_BF16(pah, bl, acc[dt]);
        acc[dt] = WMMA_BF16(pal, bh, acc[dt]);
      }
    }
#pragma unroll
    for (int dt = 0; dt < 4; ++dt)
#pragma unroll
      for (int v = 0; v < 8; ++v)
        atomicAdd(&ctx[(v + (hl ? 8 : 0)) * DH + dt * 16 + l15], acc[dt][v]);
  }
  __syncthreads();
  {
    float* cg = Out + ((size_t)b * LQ + q0) * DH;
    if (tid < QT * DH / 4)
      *(float4*)&cg[tid * 4] = *(const float4*)&ctx[tid * 4];
  }
}

extern "C" void kernel_launch(void* const* d_in, const int* in_sizes, int n_in,
                              void* d_out, int out_size, void* d_ws, size_t ws_size,
                              hipStream_t stream) {
  (void)in_sizes; (void)n_in; (void)out_size;
  const float* Q = (const float*)d_in[0];
  const float* K = (const float*)d_in[1];
  const float* V = (const float*)d_in[2];
  const int*   M = (const int*)d_in[3];
  const float* T = (const float*)d_in[4];
  float* Out = (float*)d_out;

  dim3 grid(B_ * (LQ / QT));
  dim3 block(256);

  const size_t ws_need = 4 * NKV * sizeof(bf16);      // 16 MB
  if (d_ws && ws_size >= ws_need) {
    bf16* Khi  = (bf16*)d_ws;
    bf16* Klo  = Khi + NKV;
    bf16* Vthi = Klo + NKV;
    bf16* Vtlo = Vthi + NKV;
    preprocess_kernel<<<dim3(B_ * (LK / 64)), block, 0, stream>>>(K, V, Khi, Klo, Vthi, Vtlo);
    attn_fused_kernel<true><<<grid, block, 0, stream>>>(Q, K, V, M, T, Khi, Klo, Vthi, Vtlo, Out);
  } else {
    attn_fused_kernel<false><<<grid, block, 0, stream>>>(Q, K, V, M, T,
                                                         nullptr, nullptr, nullptr, nullptr, Out);
  }
}
